// GraphResBlock_40415642255555
// MI455X (gfx1250) — matrix operand order repaired and checked by reading, not compiled, not run on hardware
//
#include <hip/hip_runtime.h>

typedef __bf16 bf16_t;
typedef __attribute__((ext_vector_type(16))) __bf16 v16bf;
typedef __attribute__((ext_vector_type(8)))  float  v8f;
typedef int v4i_gld __attribute__((vector_size(16)));  // matches builtin's int4 pointee

#define BM 128
#define BN 128
#define BK 32
#define LDT 40   // padded LDS row stride in bf16 elems (80B rows keep 16B alignment)

// ---- CDNA5 async global->LDS copy (ASYNCcnt) --------------------------------
#if __has_builtin(__builtin_amdgcn_global_load_async_to_lds_b128)
#define ASYNC_COPY_B128(dst_lds, src_glb)                                      \
  __builtin_amdgcn_global_load_async_to_lds_b128(                              \
      (__attribute__((address_space(1))) v4i_gld*)(src_glb),                   \
      (__attribute__((address_space(3))) v4i_gld*)(dst_lds), 0, 0)
#else
#define ASYNC_COPY_B128(dst_lds, src_glb)                                      \
  asm volatile("global_load_async_to_lds_b128 %0, %1, off"                     \
               :: "v"((unsigned)(unsigned long long)                           \
                      (__attribute__((address_space(3))) void*)(dst_lds)),     \
                  "v"(src_glb)                                                 \
               : "memory")
#endif

#if __has_builtin(__builtin_amdgcn_s_wait_asynccnt)
#define WAIT_ASYNC(n) __builtin_amdgcn_s_wait_asynccnt(n)
#else
#define WAIT_ASYNC(n) asm volatile("s_wait_asynccnt %0" :: "i"(n) : "memory")
#endif

// ---------------------------------------------------------------------------
// Stage a BM x BK tile into LDS as bf16.
// Fully-in-bounds bf16 tiles use async global->LDS (returns #async ops issued
// per wave = 2). Edge tiles / fp32 sources use a clamped VGPR path with
// ds_store_b128 (returns 0).
// ---------------------------------------------------------------------------
template<bool SRC_F32>
__device__ __forceinline__ int stage_tile(bf16_t* __restrict__ dst,
                                          const void* __restrict__ src,
                                          int ld, int rowBase, int rowMax,
                                          int kBase, int K, int tid)
{
  const bool fullTile = (!SRC_F32) && (rowBase + BM <= rowMax) && (kBase + BK <= K);
  if (fullTile) {
#pragma unroll
    for (int it = 0; it < 2; ++it) {
      const int c  = tid + it * 256;          // 512 chunks of 8 bf16 (16B)
      const int r  = c >> 2;
      const int k0 = (c & 3) * 8;
      const bf16_t* p = (const bf16_t*)src + (long)(rowBase + r) * ld + (kBase + k0);
      ASYNC_COPY_B128(dst + r * LDT + k0, p);
    }
    return 2;
  }
#pragma unroll
  for (int it = 0; it < 2; ++it) {
    const int c  = tid + it * 256;
    const int r  = c >> 2;
    const int k0 = (c & 3) * 8;
    const int gr = rowBase + r;
    const int gk = kBase + k0;
    const bool rowOK = (gr < rowMax);
    bf16_t tmp[8];
    if (rowOK && (gk + 8 <= K)) {
      if constexpr (SRC_F32) {
        const float* p = (const float*)src + (long)gr * ld + gk;
#pragma unroll
        for (int j = 0; j < 8; ++j) tmp[j] = (bf16_t)p[j];
      } else {
        const bf16_t* p = (const bf16_t*)src + (long)gr * ld + gk;
        *(float4*)tmp = *(const float4*)p;
      }
    } else {
      // Branch-free: clamp addresses into range, select zero for OOB lanes.
      const int  rc   = rowOK ? gr : (rowMax - 1);
      const long base = (long)rc * ld;
#pragma unroll
      for (int j = 0; j < 8; ++j) {
        const int kk = gk + j;
        const int kc = (kk < K) ? kk : (K - 1);
        float v;
        if constexpr (SRC_F32) v = ((const float*)src)[base + kc];
        else                   v = (float)((const bf16_t*)src)[base + kc];
        tmp[j] = (rowOK && kk < K) ? (bf16_t)v : (bf16_t)0.f;
      }
    }
    *(float4*)(dst + r * LDT + k0) = *(float4*)tmp;
  }
  return 0;
}

// A-matrix 16x32 bf16 fragment (lanes 0-15: K 0-7 & 16-23; lanes 16-31: K 8-15 & 24-31).
__device__ __forceinline__ v16bf frag_a(const bf16_t* s, int rowBase, int lane) {
  const int r    = rowBase + (lane & 15);
  const int koff = (lane >> 4) * 8;
  union { v16bf v; float4 f[2]; } u;
  const bf16_t* p = s + r * LDT;
  u.f[0] = *(const float4*)(p + koff);
  u.f[1] = *(const float4*)(p + koff + 16);
  return u.v;
}

// B-matrix fragment from Bt[N,K] tile: lane&15 = N column; lanes 0-15 K 0-7 & 16-23,
// lanes 16-31 K 8-15 & 24-31.
__device__ __forceinline__ v16bf frag_b(const bf16_t* s, int rowBase, int lane) {
  const int r    = rowBase + (lane & 15);
  const int koff = (lane >> 4) * 8;
  union { v16bf v; float4 f[2]; } u;
  const bf16_t* p = s + r * LDT + koff;
  u.f[0] = *(const float4*)p;
  u.f[1] = *(const float4*)(p + 16);
  return u.v;
}

// ---------------------------------------------------------------------------
// Batched GEMM: C[M,N] = A[M,K] * Bt[N,K]^T (bf16 WMMA, f32 acc), double-
// buffered LDS with async global->LDS prefetch of the next k-tile.
//   MODE 0: Cout fp32 = acc + bias[col]
//   MODE 1: Cout bf16 stored TRANSPOSED (via LDS for coalescing)
//   MODE 2: Cout fp32 = acc + bias[col] + resid[row*ldc + col]
// ---------------------------------------------------------------------------
template<bool A_F32, int MODE>
__global__ void __launch_bounds__(256)
wmma_gemm_kernel(const void* __restrict__ A, const bf16_t* __restrict__ Bt,
                 const float* __restrict__ bias, const float* __restrict__ resid,
                 void* __restrict__ Cout,
                 int M, int N, int K,
                 int lda, int ldb, int ldc,
                 long strideA, long strideB, long strideC)
{
  __shared__ bf16_t sA[2][BM * LDT];
  __shared__ bf16_t sB[2][BN * LDT];

  const int tid   = threadIdx.x;
  const int lane  = tid & 31;
  const int wave  = tid >> 5;
  const int waveM = wave & 3;   // 4 waves along M
  const int waveN = wave >> 2;  // 2 waves along N
  const int bz    = blockIdx.z;

  const void* Ab = A_F32 ? (const void*)((const float*)A + (long)bz * strideA)
                         : (const void*)((const bf16_t*)A + (long)bz * strideA);
  const bf16_t* Bb = Bt + (long)bz * strideB;

  const int rowA0 = blockIdx.y * BM;
  const int rowB0 = blockIdx.x * BN;

  const v8f vzero = {0.f, 0.f, 0.f, 0.f, 0.f, 0.f, 0.f, 0.f};
  v8f acc[2][4];
#pragma unroll
  for (int i = 0; i < 2; ++i)
#pragma unroll
    for (int j = 0; j < 4; ++j) acc[i][j] = vzero;

  const int nIter = (K + BK - 1) / BK;

  // Prologue: stage k-tile 0 into buffer 0.
  stage_tile<A_F32>(sA[0], Ab, lda, rowA0, M, 0, K, tid);
  stage_tile<false>(sB[0], (const void*)Bb, ldb, rowB0, N, 0, K, tid);

  for (int i = 0; i < nIter; ++i) {
    const int cur = i & 1;
    const int nxt = cur ^ 1;

    // Prefetch next k-tile into the other buffer (async where possible).
    int nNext = 0;
    if (i + 1 < nIter) {
      const int kb1 = (i + 1) * BK;
      nNext += stage_tile<A_F32>(sA[nxt], Ab, lda, rowA0, M, kb1, K, tid);
      nNext += stage_tile<false>(sB[nxt], (const void*)Bb, ldb, rowB0, N, kb1, K, tid);
    }
    // Async loads retire in order: allowing nNext outstanding guarantees all
    // copies for iteration i have landed in LDS. (nNext uniform per block.)
    if (nNext >= 4)      { WAIT_ASYNC(4); }
    else if (nNext >= 2) { WAIT_ASYNC(2); }
    else                 { WAIT_ASYNC(0); }
    __syncthreads();  // all waves' staging of buffer `cur` visible

    v16bf af[2], bfv[4];
#pragma unroll
    for (int ii = 0; ii < 2; ++ii) af[ii]  = frag_a(sA[cur], waveM * 32 + ii * 16, lane);
#pragma unroll
    for (int jj = 0; jj < 4; ++jj) bfv[jj] = frag_b(sB[cur], waveN * 64 + jj * 16, lane);

#pragma unroll
    for (int ii = 0; ii < 2; ++ii)
#pragma unroll
      for (int jj = 0; jj < 4; ++jj)
        acc[ii][jj] = __builtin_amdgcn_wmma_f32_16x16x32_bf16(
            false, af[ii], false, bfv[jj], (short)0, acc[ii][jj], false, false);

    __syncthreads();  // everyone done reading buffer `cur` before it's reused
  }

  // Epilogue. C/D layout: lane&15 = col, VGPR e -> row = e + (lane>>4)*8.
  const int mBase = rowA0 + waveM * 32;
  const int nBase = rowB0 + waveN * 64;

  if constexpr (MODE == 1) {
    // Transpose through LDS, then fully-coalesced 16B stores along M.
    constexpr int TP = BM + 8;  // padded row stride (272B, 16B-aligned rows)
    __shared__ bf16_t sT[BN * TP];
#pragma unroll
    for (int j = 0; j < 4; ++j) {
      const int cl = waveN * 64 + j * 16 + (lane & 15);   // local col (N dim)
#pragma unroll
      for (int i = 0; i < 2; ++i)
#pragma unroll
        for (int e = 0; e < 8; ++e) {
          const int rl = waveM * 32 + i * 16 + (lane >> 4) * 8 + e;  // local row
          sT[cl * TP + rl] = (bf16_t)acc[i][j][e];
        }
    }
    __syncthreads();
#pragma unroll
    for (int it = 0; it < 8; ++it) {
      const int c = tid + it * 256;       // 2048 chunks of 8 bf16
      const int r = c >> 4;               // local N index 0..127
      const int o = (c & 15) * 8;         // local M offset 0..120
      const int gcol = rowB0 + r;
      const int grow = rowA0 + o;
      if (gcol >= N) continue;
      bf16_t* outp = (bf16_t*)Cout + (long)bz * strideC + (long)gcol * ldc + grow;
      const bf16_t* sp = sT + r * TP + o;
      if (grow + 8 <= M) {
        *(float4*)outp = *(const float4*)sp;    // 16B coalesced store
      } else if (grow < M) {
        for (int j = 0; j < 8 && grow + j < M; ++j) outp[j] = sp[j];
      }
    }
    return;
  }

#pragma unroll
  for (int j = 0; j < 4; ++j) {
    const int col = nBase + j * 16 + (lane & 15);
    if (col >= N) continue;
    const float bv = bias[col];
#pragma unroll
    for (int i = 0; i < 2; ++i) {
#pragma unroll
      for (int e = 0; e < 8; ++e) {
        const int row = mBase + i * 16 + (lane >> 4) * 8 + e;
        if (row >= M) continue;
        float v = acc[i][j][e] + bv;
        if (MODE == 0) {
          ((float*)Cout)[(long)bz * strideC + (long)row * ldc + col] = v;
        } else {
          v += resid[(long)row * ldc + col];
          ((float*)Cout)[(long)row * ldc + col] = v;
        }
      }
    }
  }
}

// ---------------------------------------------------------------------------
// Fused LayerNorm(eps=1e-12) + ReLU + fp32->bf16. One wave32 per row.
// ---------------------------------------------------------------------------
template<int CD>
__global__ void __launch_bounds__(256)
ln_relu_kernel(const float* __restrict__ in, const float* __restrict__ gamma,
               const float* __restrict__ beta, bf16_t* __restrict__ out, int rows)
{
  constexpr int NV = CD / 128;  // float4 chunks per lane
  const int lane = threadIdx.x & 31;
  const int wave = threadIdx.x >> 5;
  const long row = (long)blockIdx.x * 8 + wave;
  if (row >= rows) return;

  const float4* p = (const float4*)(in + row * CD);
  float4 v[NV];
  float s = 0.f;
#pragma unroll
  for (int i = 0; i < NV; ++i) {
    v[i] = p[lane + i * 32];
    s += v[i].x + v[i].y + v[i].z + v[i].w;
  }
#pragma unroll
  for (int off = 16; off > 0; off >>= 1) s += __shfl_xor(s, off, 32);
  const float mean = s * (1.f / CD);

  float s2 = 0.f;
#pragma unroll
  for (int i = 0; i < NV; ++i) {
    const float dx = v[i].x - mean, dy = v[i].y - mean,
                dz = v[i].z - mean, dw = v[i].w - mean;
    s2 += dx * dx + dy * dy + dz * dz + dw * dw;
  }
#pragma unroll
  for (int off = 16; off > 0; off >>= 1) s2 += __shfl_xor(s2, off, 32);
  const float inv = rsqrtf(s2 * (1.f / CD) + 1e-12f);

#pragma unroll
  for (int i = 0; i < NV; ++i) {
    const int c4 = lane + i * 32;
    const float4 g = ((const float4*)gamma)[c4];
    const float4 b = ((const float4*)beta)[c4];
    bf16_t t[4];
    t[0] = (bf16_t)fmaxf((v[i].x - mean) * inv * g.x + b.x, 0.f);
    t[1] = (bf16_t)fmaxf((v[i].y - mean) * inv * g.y + b.y, 0.f);
    t[2] = (bf16_t)fmaxf((v[i].z - mean) * inv * g.z + b.z, 0.f);
    t[3] = (bf16_t)fmaxf((v[i].w - mean) * inv * g.w + b.w, 0.f);
    *(float2*)(out + row * CD + c4 * 4) = *(float2*)t;  // 8B store
  }
}

// Weight conversions (tiny).
__global__ void cvt_kernel(const float* __restrict__ in, bf16_t* __restrict__ out, int n) {
  const int i = blockIdx.x * 256 + threadIdx.x;
  if (i < n) out[i] = (bf16_t)in[i];
}
// in: [R,C] row-major; out: [C,R] row-major (out[c*R+r] = in[r*C+c])
__global__ void cvt_t_kernel(const float* __restrict__ in, bf16_t* __restrict__ out, int R, int C) {
  const int i = blockIdx.x * 256 + threadIdx.x;
  if (i < R * C) {
    const int r = i / C, c = i % C;
    out[c * R + r] = (bf16_t)in[i];
  }
}

extern "C" void kernel_launch(void* const* d_in, const int* in_sizes, int n_in,
                              void* d_out, int out_size, void* d_ws, size_t ws_size,
                              hipStream_t stream) {
  (void)in_sizes; (void)n_in; (void)out_size; (void)ws_size;
  const float* x      = (const float*)d_in[0];
  const float* adj    = (const float*)d_in[1];
  const float* pre_w  = (const float*)d_in[2];
  const float* pre_b  = (const float*)d_in[3];
  const float* lin1_W = (const float*)d_in[4];
  const float* lin1_b = (const float*)d_in[5];
  const float* n1_w   = (const float*)d_in[6];
  const float* n1_b   = (const float*)d_in[7];
  const float* conv_W = (const float*)d_in[8];
  const float* conv_b = (const float*)d_in[9];
  const float* n2_w   = (const float*)d_in[10];
  const float* n2_b   = (const float*)d_in[11];
  const float* lin2_W = (const float*)d_in[12];
  const float* lin2_b = (const float*)d_in[13];

  const int  Bn = 32, Nn = 1723;
  const long Mtot = (long)Bn * Nn;          // 55136
  const int  LD_ST = 1728;                  // support_t padded leading dim

  // Workspace layout (bytes, 256B aligned), with reuse: z2->y1, y3->z1.
  char* ws = (char*)d_ws;
  bf16_t* w1  = (bf16_t*)(ws + 0);                      // lin1_W bf16 [256,512]
  bf16_t* w2  = (bf16_t*)(ws + 262144);                 // lin2_W bf16 [512,256]
  bf16_t* wct = (bf16_t*)(ws + 524288);                 // conv_W^T bf16 [256,256]
  bf16_t* y1  = (bf16_t*)(ws + 655360);                 // [55136,512] bf16
  float*  z1  = (float* )(ws + 655360 + 56459264L);     // [55136,256] f32
  bf16_t* y2  = (bf16_t*)(ws + 655360 + 112918528L);    // [55136,256] bf16
  bf16_t* st  = (bf16_t*)(ws + 655360 + 141148160L);    // [32,256,1728] bf16
  float*  z2  = (float*)y1;                             // reuse
  bf16_t* y3  = (bf16_t*)z1;                            // reuse

  // Weight prep
  cvt_kernel<<<512, 256, 0, stream>>>(lin1_W, w1, 131072);
  cvt_kernel<<<512, 256, 0, stream>>>(lin2_W, w2, 131072);
  cvt_t_kernel<<<256, 256, 0, stream>>>(conv_W, wct, 256, 256);

  const int lnBlocks = (int)((Mtot + 7) / 8);
  const int gy       = (int)((Mtot + BM - 1) / BM);  // 431
  const int gyB      = (Nn + BM - 1) / BM;           // 14

  // 1) pre_norm + relu -> bf16
  ln_relu_kernel<512><<<lnBlocks, 256, 0, stream>>>(x, pre_w, pre_b, y1, (int)Mtot);

  // 2) lin1: y1[M,512] x lin1_W[256,512]^T + b -> z1 f32 [M,256]
  wmma_gemm_kernel<false, 0><<<dim3(2, gy, 1), 256, 0, stream>>>(
      y1, w1, lin1_b, nullptr, z1, (int)Mtot, 256, 512, 512, 512, 256, 0, 0, 0);

  // 3) LN + relu -> y2 bf16
  ln_relu_kernel<256><<<lnBlocks, 256, 0, stream>>>(z1, n1_w, n1_b, y2, (int)Mtot);

  // 4) support = y2 x conv_W -> stored transposed per batch: st[b][k][m]
  wmma_gemm_kernel<false, 1><<<dim3(2, gyB, 32), 256, 0, stream>>>(
      y2, wct, nullptr, nullptr, st, Nn, 256, 256, 256, 256, LD_ST,
      (long)Nn * 256, 0, (long)256 * LD_ST);

  // 5) adj GEMM: adj(f32, converted on the fly) x st^T + conv_b -> z2 f32
  wmma_gemm_kernel<true, 0><<<dim3(2, gyB, 32), 256, 0, stream>>>(
      adj, st, conv_b, nullptr, z2, Nn, 256, Nn, Nn, LD_ST, 256,
      (long)Nn * Nn, (long)256 * LD_ST, (long)Nn * 256);

  // 6) LN + relu -> y3 bf16
  ln_relu_kernel<256><<<lnBlocks, 256, 0, stream>>>(z2, n2_w, n2_b, y3, (int)Mtot);

  // 7) lin2 + bias + residual(x) -> d_out f32
  wmma_gemm_kernel<false, 2><<<dim3(4, gy, 1), 256, 0, stream>>>(
      y3, w2, lin2_b, x, d_out, (int)Mtot, 512, 256, 256, 256, 512, 0, 0, 0);
}
